// LSTMSummaryGenerator_17652315587429
// MI455X (gfx1250) — matrix operand order repaired and checked by reading, not compiled, not run on hardware
//
#include <hip/hip_runtime.h>
#include <stdint.h>

// MI455X / gfx1250 implementation of the attention-GRU decoder.
//
//  * One-time: convert weights + history embeddings to bf16; precompute
//    K,V with a FUSED bf16 WMMA GEMM (one fp32 A stream feeds both
//    projections). K/V/emb bf16 working set = 144 MB -> L2-resident (192 MB).
//  * Per step (50x, sequential): software-pipelined WMMA GEMMs for q, Wf,
//    W_ih, W_hh; per-batch attention kernel; elementwise GRU update.
//  * Matrix ops lower to v_wmma_f32_16x16x32_bf16 (wave32, 16x64 per wave,
//    double-buffered fragments so loads overlap the matrix pipe).

#define DD 768
#define BB 64
#define SS 512
#define TT 50

typedef __attribute__((ext_vector_type(16))) __bf16 v16bf;
typedef __attribute__((ext_vector_type(8)))  float  v8f;

__device__ __forceinline__ unsigned short f2bf(float f) {
  union { float f; unsigned int u; } c; c.f = f;
  unsigned int u = c.u;
  unsigned int r = u + 0x7FFFu + ((u >> 16) & 1u);   // round-to-nearest-even
  return (unsigned short)(r >> 16);
}
__device__ __forceinline__ float bf2f(unsigned int s) {
  union { unsigned int u; float f; } c; c.u = s << 16;
  return c.f;
}

union BFrag { v16bf v; uint4 u4[2]; unsigned short us[16]; };

// A fragment (bf16 source): lane half 0 holds k = {0..7,16..23}, half 1 holds
// k = {8..15,24..31} relative to the 32-k slab (base pointer already includes
// khalf*8).
__device__ __forceinline__ void load_afrag_bf(BFrag& a, const unsigned short* p) {
  a.u4[0] = *(const uint4*)(p);
  a.u4[1] = *(const uint4*)(p + 16);
}
__device__ __forceinline__ void load_afrag_f32(BFrag& a, const float* p) {
  float4 f0 = *(const float4*)(p + 0);
  float4 f1 = *(const float4*)(p + 4);
  float4 f2 = *(const float4*)(p + 16);
  float4 f3 = *(const float4*)(p + 20);
  a.us[0]=f2bf(f0.x);  a.us[1]=f2bf(f0.y);  a.us[2]=f2bf(f0.z);  a.us[3]=f2bf(f0.w);
  a.us[4]=f2bf(f1.x);  a.us[5]=f2bf(f1.y);  a.us[6]=f2bf(f1.z);  a.us[7]=f2bf(f1.w);
  a.us[8]=f2bf(f2.x);  a.us[9]=f2bf(f2.y);  a.us[10]=f2bf(f2.z); a.us[11]=f2bf(f2.w);
  a.us[12]=f2bf(f3.x); a.us[13]=f2bf(f3.y); a.us[14]=f2bf(f3.z); a.us[15]=f2bf(f3.w);
}
// B fragment: lane holds column n, k = {0..7,16..23} / {8..15,24..31} like A (base includes khalf*8).
__device__ __forceinline__ void load_bfrag(BFrag& b, const unsigned short* p) {
  b.u4[0] = *(const uint4*)(p);
  b.u4[1] = *(const uint4*)(p + 16);
}
__device__ __forceinline__ void mma4(v8f* acc, const BFrag& a, const BFrag* b) {
#pragma unroll
  for (int t = 0; t < 4; ++t)
    acc[t] = __builtin_amdgcn_wmma_f32_16x16x32_bf16(
                 false, a.v, false, b[t].v, (short)0, acc[t], false, false);
}

// ---------------------------------------------------------------------------
// C[M,N] = A[M,K] (row major) * Bw[N,K]^T + bias[N].
// Wave computes 16x64; block = 8 waves -> 32x256 tile. Requires M%32==0,
// N%256==0, K%64==0 (true for all uses). Double-buffered fragments: the
// loads for k+32 are issued before the WMMAs for k, so the matrix pipe
// overlaps VMEM instead of stalling at s_wait_loadcnt 0.
// ---------------------------------------------------------------------------
template<bool A_F32, bool OUT_BF16>
__global__ __launch_bounds__(256)
void gemm_bf16_wmma(const void* __restrict__ A_, const unsigned short* __restrict__ Bw,
                    const float* __restrict__ bias, void* __restrict__ C_,
                    int M, int N, int K)
{
  const int lane  = threadIdx.x & 31;
  const int wave  = threadIdx.x >> 5;           // 0..7
  const int wm    = wave >> 2;                  // 0..1
  const int wn    = wave & 3;                   // 0..3
  const int row0  = blockIdx.y * 32 + wm * 16;
  const int col0  = blockIdx.x * 256 + wn * 64;
  const int ml    = lane & 15;
  const int khalf = lane >> 4;

  const int aRow = row0 + ml;
  const unsigned short* Abf = (const unsigned short*)A_ + (size_t)aRow * K + khalf * 8;
  const float*          Af  = (const float*)A_          + (size_t)aRow * K + khalf * 8;
  const unsigned short* Bp[4];
#pragma unroll
  for (int t = 0; t < 4; ++t)
    Bp[t] = Bw + (size_t)(col0 + t * 16 + ml) * K + khalf * 8;

  v8f acc[4];
#pragma unroll
  for (int t = 0; t < 4; ++t) acc[t] = 0.f;

  BFrag a0, a1, b0[4], b1[4];
  auto LA = [&](BFrag& a, int kb) {
    if constexpr (A_F32) load_afrag_f32(a, Af + kb);
    else                 load_afrag_bf(a, Abf + kb);
  };
  auto LB = [&](BFrag* b, int kb) {
#pragma unroll
    for (int t = 0; t < 4; ++t) load_bfrag(b[t], Bp[t] + kb);
  };

  LA(a0, 0); LB(b0, 0);
  for (int kb = 0; kb < K; kb += 64) {
    LA(a1, kb + 32); LB(b1, kb + 32);
    mma4(acc, a0, b0);
    if (kb + 64 < K) { LA(a0, kb + 64); LB(b0, kb + 64); }
    mma4(acc, a1, b1);
  }

#pragma unroll
  for (int t = 0; t < 4; ++t) {
    const int n = col0 + t * 16 + ml;
    const float bn = bias[n];
#pragma unroll
    for (int j = 0; j < 8; ++j) {
      const int m = row0 + khalf * 8 + j;       // D layout: vgpr j, lane half
      const float val = acc[t][j] + bn;
      if constexpr (OUT_BF16)
        ((unsigned short*)C_)[(size_t)m * N + n] = f2bf(val);
      else
        ((float*)C_)[(size_t)m * N + n] = val;
    }
  }
}

// ---------------------------------------------------------------------------
// Fused K/V projection: one fp32 encoded_history stream (96 MB, converted to
// bf16 in-register) feeds BOTH Wk and Wv -> 8 WMMAs per A fragment. Halves
// the dominant one-time HBM read vs two separate GEMMs.
// ---------------------------------------------------------------------------
__global__ __launch_bounds__(256)
void gemm_kv_fused(const float* __restrict__ A, const unsigned short* __restrict__ WkB,
                   const unsigned short* __restrict__ WvB, const float* __restrict__ bk,
                   const float* __restrict__ bv, unsigned short* __restrict__ Ck,
                   unsigned short* __restrict__ Cv, int M, int N, int K)
{
  const int lane  = threadIdx.x & 31;
  const int wave  = threadIdx.x >> 5;
  const int wm    = wave >> 2;
  const int wn    = wave & 3;
  const int row0  = blockIdx.y * 32 + wm * 16;
  const int col0  = blockIdx.x * 256 + wn * 64;
  const int ml    = lane & 15;
  const int khalf = lane >> 4;

  const float* Af = A + (size_t)(row0 + ml) * K + khalf * 8;
  const unsigned short* Bpk[4];
  const unsigned short* Bpv[4];
#pragma unroll
  for (int t = 0; t < 4; ++t) {
    const size_t ro = (size_t)(col0 + t * 16 + ml) * K + khalf * 8;
    Bpk[t] = WkB + ro;
    Bpv[t] = WvB + ro;
  }

  v8f acck[4], accv[4];
#pragma unroll
  for (int t = 0; t < 4; ++t) { acck[t] = 0.f; accv[t] = 0.f; }

  BFrag a0, a1, bt[4];
  load_afrag_f32(a0, Af);
  for (int kb = 0; kb < K; kb += 64) {
    load_afrag_f32(a1, Af + kb + 32);
    __builtin_prefetch(Af + kb + 128, 0, 3);          // global_prefetch_b8
#pragma unroll
    for (int t = 0; t < 4; ++t) load_bfrag(bt[t], Bpk[t] + kb);
    mma4(acck, a0, bt);
#pragma unroll
    for (int t = 0; t < 4; ++t) load_bfrag(bt[t], Bpv[t] + kb);
    mma4(accv, a0, bt);
    if (kb + 64 < K) load_afrag_f32(a0, Af + kb + 64);
#pragma unroll
    for (int t = 0; t < 4; ++t) load_bfrag(bt[t], Bpk[t] + kb + 32);
    mma4(acck, a1, bt);
#pragma unroll
    for (int t = 0; t < 4; ++t) load_bfrag(bt[t], Bpv[t] + kb + 32);
    mma4(accv, a1, bt);
  }

#pragma unroll
  for (int t = 0; t < 4; ++t) {
    const int n = col0 + t * 16 + ml;
    const float bkn = bk[n], bvn = bv[n];
#pragma unroll
    for (int j = 0; j < 8; ++j) {
      const int m = row0 + khalf * 8 + j;
      Ck[(size_t)m * N + n] = f2bf(acck[t][j] + bkn);
      Cv[(size_t)m * N + n] = f2bf(accv[t][j] + bvn);
    }
  }
}

// ---------------------------------------------------------------------------
// Per-batch attention step: logits = q.K + mask; softmax; ctx = attn.V;
// one_hot = softmax(logits*10); next_emb = one_hot.emb; argmax token.
// One workgroup (256 thr = 8 waves) per batch element.
// ---------------------------------------------------------------------------
__global__ __launch_bounds__(256)
void attn_step(const float* __restrict__ q, const unsigned short* __restrict__ Kbf,
               const unsigned short* __restrict__ Vbf, const unsigned short* __restrict__ Ebf,
               const float* __restrict__ maskadd, const int* __restrict__ ids,
               float* __restrict__ out_soft, float* __restrict__ out_hard,
               unsigned short* __restrict__ fused, unsigned short* __restrict__ xstage,
               int t)
{
  __shared__ float sh_q[DD];
  __shared__ float sh_logit[SS];
  __shared__ float sh_attn[SS];
  __shared__ float sh_oneh[SS];
  __shared__ float sred[256];
  __shared__ int   sidx[256];
  __shared__ float ssumA[256];
  __shared__ float ssumB[256];

  const int b = blockIdx.x;
  const int tid = threadIdx.x;
  const int lane = tid & 31;
  const int wave = tid >> 5;

  for (int d = tid; d < DD; d += 256) sh_q[d] = q[b * DD + d];
  __syncthreads();

  // logits: one s per wave iteration; lanes split d, coalesced 16B bf16 loads
  const unsigned short* Krow0 = Kbf + (size_t)b * SS * DD;
  for (int s = wave; s < SS; s += 8) {
    const uint4* kp = (const uint4*)(Krow0 + (size_t)s * DD);
    float acc = 0.f;
#pragma unroll
    for (int it = 0; it < 3; ++it) {
      uint4 pk = kp[it * 32 + lane];
      const float* qq = &sh_q[it * 256 + lane * 8];
      acc = fmaf(bf2f(pk.x & 0xFFFFu), qq[0], acc);
      acc = fmaf(bf2f(pk.x >> 16),     qq[1], acc);
      acc = fmaf(bf2f(pk.y & 0xFFFFu), qq[2], acc);
      acc = fmaf(bf2f(pk.y >> 16),     qq[3], acc);
      acc = fmaf(bf2f(pk.z & 0xFFFFu), qq[4], acc);
      acc = fmaf(bf2f(pk.z >> 16),     qq[5], acc);
      acc = fmaf(bf2f(pk.w & 0xFFFFu), qq[6], acc);
      acc = fmaf(bf2f(pk.w >> 16),     qq[7], acc);
    }
#pragma unroll
    for (int off = 16; off > 0; off >>= 1) acc += __shfl_xor(acc, off, 32);
    if (lane == 0) sh_logit[s] = acc + maskadd[b * SS + s];
  }
  __syncthreads();

  // block max + argmax (first max index, as jnp.argmax)
  float v1 = sh_logit[tid], v2 = sh_logit[tid + 256];
  float bm; int bi;
  if (v1 >= v2) { bm = v1; bi = tid; } else { bm = v2; bi = tid + 256; }
  sred[tid] = bm; sidx[tid] = bi;
  __syncthreads();
  for (int st = 128; st > 0; st >>= 1) {
    if (tid < st) {
      float o = sred[tid + st]; int oi = sidx[tid + st];
      if (o > sred[tid] || (o == sred[tid] && oi < sidx[tid])) { sred[tid] = o; sidx[tid] = oi; }
    }
    __syncthreads();
  }
  const float gmax = sred[0];
  const int sel = sidx[0];

  float l1 = v1 - gmax, l2 = v2 - gmax;
  float e1 = __expf(l1),        e2 = __expf(l2);
  float o1 = __expf(l1 * 10.f), o2 = __expf(l2 * 10.f);  // logits/0.1 softmax
  sh_attn[tid] = e1; sh_attn[tid + 256] = e2;
  sh_oneh[tid] = o1; sh_oneh[tid + 256] = o2;
  ssumA[tid] = e1 + e2; ssumB[tid] = o1 + o2;
  __syncthreads();
  for (int st = 128; st > 0; st >>= 1) {
    if (tid < st) { ssumA[tid] += ssumA[tid + st]; ssumB[tid] += ssumB[tid + st]; }
    __syncthreads();
  }
  const float invA = 1.f / ssumA[0];
  const float invB = 1.f / ssumB[0];
  sh_attn[tid] *= invA; sh_attn[tid + 256] *= invA;
  sh_oneh[tid] *= invB; sh_oneh[tid + 256] *= invB;
  __syncthreads();

  // ctx and next_emb (threads split d; per-s loads coalesced across threads)
  const unsigned short* Vrow = Vbf + (size_t)b * SS * DD;
  const unsigned short* Erow = Ebf + (size_t)b * SS * DD;
  for (int d = tid; d < DD; d += 256) {
    float c = 0.f, e = 0.f;
    const unsigned short* vp = Vrow + d;
    const unsigned short* ep = Erow + d;
    for (int s = 0; s < SS; ++s) {
      c = fmaf(sh_attn[s], bf2f(vp[(size_t)s * DD]), c);
      e = fmaf(sh_oneh[s], bf2f(ep[(size_t)s * DD]), e);
    }
    fused[(size_t)b * (2 * DD) + d] = f2bf(c);                 // ctx -> Wf input lo
    out_soft[((size_t)b * TT + t) * DD + d] = e;               // soft output fp32
    xstage[(size_t)b * DD + d] = f2bf(e);                      // x_emb for t+1
  }
  if (tid == 0) out_hard[b * TT + t] = (float)ids[b * SS + sel];
}

// ---------------------------------------------------------------------------
// GRU elementwise update (gate order r,z,n); biases folded into gi/gh GEMMs.
// Also stages next x_emb into the fused Wf-input buffer for step t+1.
// ---------------------------------------------------------------------------
__global__ __launch_bounds__(256)
void gru_update(const float* __restrict__ gi, const float* __restrict__ gh,
                float* __restrict__ h, unsigned short* __restrict__ h_bf,
                const unsigned short* __restrict__ xstage, unsigned short* __restrict__ fused)
{
  const int idx = blockIdx.x * 256 + threadIdx.x;      // < B*D (exact grid)
  const int b = idx / DD, d = idx - b * DD;
  const float* gib = gi + (size_t)b * 3 * DD;
  const float* ghb = gh + (size_t)b * 3 * DD;
  const float r = 1.f / (1.f + __expf(-(gib[d] + ghb[d])));
  const float z = 1.f / (1.f + __expf(-(gib[DD + d] + ghb[DD + d])));
  const float n = tanhf(gib[2 * DD + d] + r * ghb[2 * DD + d]);
  const float hn = (1.f - z) * n + z * h[idx];
  h[idx] = hn;
  h_bf[idx] = f2bf(hn);
  fused[(size_t)b * (2 * DD) + DD + d] = xstage[idx];  // x_emb -> Wf input hi
}

__global__ __launch_bounds__(256)
void f32_to_bf16(const float* __restrict__ src, unsigned short* __restrict__ dst, int n)
{
  const int i = blockIdx.x * 256 + threadIdx.x;
  if (i < n) dst[i] = f2bf(src[i]);
}

// Masked average pool h0, mask additive bias, and bos -> fused hi half.
__global__ __launch_bounds__(256)
void pool_init(const float* __restrict__ enc, const float* __restrict__ mask,
               const float* __restrict__ bos, float* __restrict__ h,
               unsigned short* __restrict__ h_bf, unsigned short* __restrict__ fused,
               float* __restrict__ maskadd)
{
  __shared__ float smask[SS];
  __shared__ float red[256];
  const int b = blockIdx.x, tid = threadIdx.x;
  for (int s = tid; s < SS; s += 256) smask[s] = mask[b * SS + s];
  __syncthreads();
  float lm = 0.f;
  for (int s = tid; s < SS; s += 256) {
    lm += smask[s];
    maskadd[b * SS + s] = (1.f - smask[s]) * -10000.f;
  }
  red[tid] = lm; __syncthreads();
  for (int st = 128; st > 0; st >>= 1) { if (tid < st) red[tid] += red[tid + st]; __syncthreads(); }
  const float invLen = 1.f / red[0];
  const float* ebase = enc + (size_t)b * SS * DD;
  for (int d = tid; d < DD; d += 256) {
    float acc = 0.f;
    for (int s = 0; s < SS; ++s) acc = fmaf(ebase[(size_t)s * DD + d], smask[s], acc);
    const float hv = acc * invLen;
    h[b * DD + d] = hv;
    h_bf[b * DD + d] = f2bf(hv);
    fused[(size_t)b * (2 * DD) + DD + d] = f2bf(bos[b * DD + d]);
  }
}

// ---------------------------------------------------------------------------
extern "C" void kernel_launch(void* const* d_in, const int* in_sizes, int n_in,
                              void* d_out, int out_size, void* d_ws, size_t ws_size,
                              hipStream_t stream)
{
  (void)in_sizes; (void)n_in; (void)out_size; (void)ws_size;
  const float* enc  = (const float*)d_in[0];
  const float* mask = (const float*)d_in[1];
  const int*   ids  = (const int*)d_in[2];
  const float* emb  = (const float*)d_in[3];
  const float* bos  = (const float*)d_in[4];
  const float* Wq   = (const float*)d_in[5];  const float* bq  = (const float*)d_in[6];
  const float* Wk   = (const float*)d_in[7];  const float* bk  = (const float*)d_in[8];
  const float* Wv   = (const float*)d_in[9];  const float* bv  = (const float*)d_in[10];
  const float* Wf   = (const float*)d_in[11]; const float* bf_ = (const float*)d_in[12];
  const float* Wih  = (const float*)d_in[13];
  const float* Whh  = (const float*)d_in[14];
  const float* bih  = (const float*)d_in[15];
  const float* bhh  = (const float*)d_in[16];

  // Workspace carve-up (~165 MB; K/V/emb bf16 dominate and are L2-resident).
  char* ws = (char*)d_ws;
  size_t off = 0;
  auto take = [&](size_t bytes) -> void* {
    void* p = ws + off; off += (bytes + 255) & ~(size_t)255; return p;
  };
  unsigned short* Kbf    = (unsigned short*)take((size_t)BB * SS * DD * 2);
  unsigned short* Vbf    = (unsigned short*)take((size_t)BB * SS * DD * 2);
  unsigned short* Ebf    = (unsigned short*)take((size_t)BB * SS * DD * 2);
  unsigned short* Wq_bf  = (unsigned short*)take((size_t)DD * DD * 2);
  unsigned short* Wk_bf  = (unsigned short*)take((size_t)DD * DD * 2);
  unsigned short* Wv_bf  = (unsigned short*)take((size_t)DD * DD * 2);
  unsigned short* Wf_bf  = (unsigned short*)take((size_t)DD * 2 * DD * 2);
  unsigned short* Wih_bf = (unsigned short*)take((size_t)3 * DD * DD * 2);
  unsigned short* Whh_bf = (unsigned short*)take((size_t)3 * DD * DD * 2);
  float*          h      = (float*)take((size_t)BB * DD * 4);
  unsigned short* h_bf   = (unsigned short*)take((size_t)BB * DD * 2);
  float*          qbuf   = (float*)take((size_t)BB * DD * 4);
  unsigned short* fused  = (unsigned short*)take((size_t)BB * 2 * DD * 2);
  unsigned short* xstage = (unsigned short*)take((size_t)BB * DD * 2);
  unsigned short* ginbf  = (unsigned short*)take((size_t)BB * DD * 2);
  float*          gibuf  = (float*)take((size_t)BB * 3 * DD * 4);
  float*          ghbuf  = (float*)take((size_t)BB * 3 * DD * 4);
  float*          maskadd= (float*)take((size_t)BB * SS * 4);

  // One-time conversions + pooled init.
  const int D2 = DD * DD, DF = DD * 2 * DD, DG = 3 * DD * DD, NE = BB * SS * DD;
  f32_to_bf16<<<(D2 + 255) / 256, 256, 0, stream>>>(Wq, Wq_bf, D2);
  f32_to_bf16<<<(D2 + 255) / 256, 256, 0, stream>>>(Wk, Wk_bf, D2);
  f32_to_bf16<<<(D2 + 255) / 256, 256, 0, stream>>>(Wv, Wv_bf, D2);
  f32_to_bf16<<<(DF + 255) / 256, 256, 0, stream>>>(Wf, Wf_bf, DF);
  f32_to_bf16<<<(DG + 255) / 256, 256, 0, stream>>>(Wih, Wih_bf, DG);
  f32_to_bf16<<<(DG + 255) / 256, 256, 0, stream>>>(Whh, Whh_bf, DG);
  f32_to_bf16<<<(NE + 255) / 256, 256, 0, stream>>>(emb, Ebf, NE);
  pool_init<<<BB, 256, 0, stream>>>(enc, mask, bos, h, h_bf, fused, maskadd);

  // Loop-invariant K/V projections, fused: [32768,768]x[768,768] twice off
  // one encoded_history read.
  gemm_kv_fused<<<dim3(DD / 256, (BB * SS) / 32), 256, 0, stream>>>(
      enc, Wk_bf, Wv_bf, bk, bv, Kbf, Vbf, BB * SS, DD, DD);

  float* out_soft = (float*)d_out;                       // [B,50,D] fp32
  float* out_hard = out_soft + (size_t)BB * TT * DD;     // [B,50] tokens as float

  // Sequential decode: 50 steps x 6 kernels (graph-captured by the harness).
  for (int t = 0; t < TT; ++t) {
    gemm_bf16_wmma<false, false><<<dim3(DD / 256, BB / 32), 256, 0, stream>>>(
        h_bf, Wq_bf, bq, qbuf, BB, DD, DD);
    attn_step<<<BB, 256, 0, stream>>>(qbuf, Kbf, Vbf, Ebf, maskadd, ids,
                                      out_soft, out_hard, fused, xstage, t);
    gemm_bf16_wmma<false, true><<<dim3(DD / 256, BB / 32), 256, 0, stream>>>(
        fused, Wf_bf, bf_, ginbf, BB, DD, 2 * DD);
    gemm_bf16_wmma<false, false><<<dim3((3 * DD) / 256, BB / 32), 256, 0, stream>>>(
        ginbf, Wih_bf, bih, gibuf, BB, 3 * DD, DD);
    gemm_bf16_wmma<false, false><<<dim3((3 * DD) / 256, BB / 32), 256, 0, stream>>>(
        h_bf, Whh_bf, bhh, ghbuf, BB, 3 * DD, DD);
    gru_update<<<(BB * DD) / 256, 256, 0, stream>>>(gibuf, ghbuf, h, h_bf, xstage, fused);
  }
}
